// customGraphSAGE4_29154238005823
// MI455X (gfx1250) — compile-verified, matrix operand order checked by reading, not run on hardware
//
#include <hip/hip_runtime.h>
#include <hip/hip_bf16.h>

#define DEV_INLINE __device__ __forceinline__

typedef __attribute__((ext_vector_type(16))) __bf16 v16bf;
typedef __attribute__((ext_vector_type(8)))  float  v8f;

namespace {
constexpr int kE  = 16384;   // edges
constexpr int kNB = 10240;   // nodes in batch (B*N)
constexpr int kN  = 40;      // nodes per graph
constexpr int kZ  = 38;      // N - 2
constexpr int kWPB = 4;      // waves per block for WMMA kernels
}

// ---------------------------------------------------------------------------
// CDNA5 16-bit WMMA fragment index helpers (ISA 7.12.2 layouts, wave32).
// A (16x32, bf16): lane = m + 16*((kk>>3)&1), per-lane slot idx:
//   idx 0..7  -> kk = idx           (+8 group offset for lanes>=16)
//   idx 8..15 -> kk = 16 + (idx&7)
// B (32x16) is the same mapping with n in place of m.
// C/D (f32 16x16): vgpr r, lane l -> m = r + 8*(l>=16), n = l&15.
// ---------------------------------------------------------------------------

DEV_INLINE void row_decode(int r, const int* __restrict__ ei,
                           int& e, int& zi, int& zg) {
  e  = r / kZ;
  zi = r - e * kZ;
  const int v = ei[e];
  const int u = ei[kE + e];
  const int g  = v / kN;
  const int vl = v - g * kN;
  const int ul = u - g * kN;
  const int a = vl < ul ? vl : ul;
  const int b = vl < ul ? ul : vl;
  int z = zi;
  if (z >= a) ++z;
  if (z >= b) ++z;              // ascending locals excluding v,u
  zg = g * kN + z;
}

// ---------------------------------------------------------------------------
// Shared MLP tail: rows[16] x (256 ->relu-> 128 ->relu-> 64 -> dot W4 + b4)
// afrag: 8 A-fragments (K=256) already in WMMA layout.
// Returns per-row scalar; valid on lanes with (lane&1)==0 for row m=lane>>1.
// ---------------------------------------------------------------------------
DEV_INLINE float mlp_tail_256_128_64(
    const v16bf* afrag,
    const __bf16* __restrict__ wfL2, const float* __restrict__ b2,
    const __bf16* __restrict__ wfL3, const float* __restrict__ b3,
    const float* __restrict__ W4,   const float* __restrict__ b4,
    __bf16* aL3, float* a3, int lane)
{
  const int nl = lane & 15;
  const int mh = (lane >> 4) << 3;   // C-fragment row half: 0 or 8

  // ---- layer 2: [16x256] @ [256x128], bias+relu, emit K=128 A-fragments ----
#pragma unroll
  for (int nt = 0; nt < 8; ++nt) {
    v8f acc = {0.f, 0.f, 0.f, 0.f, 0.f, 0.f, 0.f, 0.f};
#pragma unroll
    for (int c = 0; c < 8; ++c) {
      const v16bf b = *(const v16bf*)(wfL2 + (size_t)((c * 8 + nt) * 32 + lane) * 16);
      acc = __builtin_amdgcn_wmma_f32_16x16x32_bf16(false, afrag[c], false, b,
                                                    (short)0, acc, false, false);
    }
    const int   j    = nt * 16 + nl;
    const float bias = b2[j];
    const int c2  = j >> 5;
    const int kk  = j & 31;
    const int lad = ((kk >> 3) & 1) * 16;
    const int idx = (kk & 7) + ((kk >> 4) & 1) * 8;
#pragma unroll
    for (int r = 0; r < 8; ++r) {
      const int m = mh + r;
      float v = acc[r] + bias;
      v = v > 0.f ? v : 0.f;
      aL3[(size_t)(c2 * 32 + m + lad) * 16 + idx] = (__bf16)v;
    }
  }
  asm volatile("s_wait_dscnt 0" ::: "memory");

  // ---- layer 3: [16x128] @ [128x64], bias+relu -> a3[16][64] f32 ----------
#pragma unroll
  for (int nt = 0; nt < 4; ++nt) {
    v8f acc = {0.f, 0.f, 0.f, 0.f, 0.f, 0.f, 0.f, 0.f};
#pragma unroll
    for (int c = 0; c < 4; ++c) {
      const v16bf a = *(const v16bf*)(aL3 + (size_t)(c * 32 + lane) * 16);
      const v16bf b = *(const v16bf*)(wfL3 + (size_t)((c * 4 + nt) * 32 + lane) * 16);
      acc = __builtin_amdgcn_wmma_f32_16x16x32_bf16(false, a, false, b,
                                                    (short)0, acc, false, false);
    }
    const int   j    = nt * 16 + nl;
    const float bias = b3[j];
#pragma unroll
    for (int r = 0; r < 8; ++r) {
      float v = acc[r] + bias;
      v = v > 0.f ? v : 0.f;
      a3[(mh + r) * 64 + j] = v;
    }
  }
  asm volatile("s_wait_dscnt 0" ::: "memory");

  // ---- layer 4: dot(a3[m], W4) + b4, 2 lanes per row -----------------------
  const int m4   = lane >> 1;
  const int half = lane & 1;
  float s = 0.f;
#pragma unroll
  for (int k = 0; k < 32; ++k)
    s = fmaf(a3[m4 * 64 + half * 32 + k], W4[half * 32 + k], s);
  s += __shfl_xor(s, 1, 32);
  return s + b4[0];
}

// ---------------------------------------------------------------------------
// Prep: re-pack f32 weight matrix [K][Nf] into bf16 B-fragment order
// ---------------------------------------------------------------------------
__global__ __launch_bounds__(256) void wfrag_kernel(
    const float* __restrict__ W, __bf16* __restrict__ out, int K, int Nf)
{
  const int t = blockIdx.x * blockDim.x + threadIdx.x;
  if (t >= K * Nf) return;
  const int k = t / Nf, n = t - (t / Nf) * Nf;
  const int c = k >> 5, kk = k & 31;
  const int nt = n >> 4, nlc = n & 15;
  const int NT = Nf >> 4;
  const int lane = nlc + ((kk >> 3) & 1) * 16;
  const int idx  = (kk & 7) + ((kk >> 4) & 1) * 8;
  out[(size_t)((c * NT + nt) * 32 + lane) * 16 + idx] = (__bf16)W[(size_t)k * Nf + n];
}

// ---------------------------------------------------------------------------
// SAGE conv: message GEMM + atomic mean-aggregation into destination u
// ---------------------------------------------------------------------------
__global__ __launch_bounds__(256) void conv_msg_kernel(
    const float* __restrict__ x, const int* __restrict__ ei,
    const float* __restrict__ ef, const float* __restrict__ Wm,
    const float* __restrict__ bm, float* __restrict__ agg, float* __restrict__ cnt)
{
  const int t = blockIdx.x * blockDim.x + threadIdx.x;
  if (t >= kE * 64) return;
  const int e = t >> 6, dd = t & 63;
  const int v = ei[e], u = ei[kE + e];
  float s = bm[dd];
  const float* xv = x + (size_t)v * 64;
#pragma unroll
  for (int k = 0; k < 64; ++k) s = fmaf(xv[k], Wm[k * 64 + dd], s);
  const float* efr = ef + (size_t)e * 16;
#pragma unroll
  for (int k = 0; k < 16; ++k) s = fmaf(efr[k], Wm[(64 + k) * 64 + dd], s);
  atomicAdd(&agg[(size_t)u * 64 + dd], s);
  if (dd == 0) atomicAdd(&cnt[u], 1.0f);
}

__global__ __launch_bounds__(256) void node_h_kernel(
    const float* __restrict__ x, const float* __restrict__ Wr,
    const float* __restrict__ br, const float* __restrict__ agg,
    const float* __restrict__ cnt, float* __restrict__ h)
{
  const int t = blockIdx.x * blockDim.x + threadIdx.x;
  if (t >= kNB * 64) return;
  const int n = t >> 6, dd = t & 63;
  float s = br[dd];
  const float* xn = x + (size_t)n * 64;
#pragma unroll
  for (int k = 0; k < 64; ++k) s = fmaf(xn[k], Wr[k * 64 + dd], s);
  float c = cnt[n];
  c = c > 1.0f ? c : 1.0f;
  s += agg[t] / c;
  h[t] = s > 0.f ? s : 0.f;
}

// tz[n,j] = h[n,:] . W2_1[128+.., j]   (per-node part of MLP2 layer 1)
__global__ __launch_bounds__(256) void tz_kernel(
    const float* __restrict__ h, const float* __restrict__ W21,
    float* __restrict__ tz)
{
  const int t = blockIdx.x * blockDim.x + threadIdx.x;
  if (t >= kNB * 256) return;
  const int n = t >> 8, j = t & 255;
  float s = 0.f;
  const float* hn = h + (size_t)n * 64;
#pragma unroll
  for (int k = 0; k < 64; ++k) s = fmaf(hn[k], W21[(128 + k) * 256 + j], s);
  tz[t] = s;
}

// out256[e,j] = bias[j] + hv.W[0:64] + hu.W[64:128] + ef.W[efRow:efRow+16]
__global__ __launch_bounds__(256) void edge_affine_kernel(
    const float* __restrict__ h, const int* __restrict__ ei,
    const float* __restrict__ ef, const float* __restrict__ W,
    const float* __restrict__ bias, int efRow, int doRelu,
    float* __restrict__ out256)
{
  const int t = blockIdx.x * blockDim.x + threadIdx.x;
  if (t >= kE * 256) return;
  const int e = t >> 8, j = t & 255;
  const int v = ei[e], u = ei[kE + e];
  float s = bias[j];
  const float* hv = h + (size_t)v * 64;
  const float* hu = h + (size_t)u * 64;
#pragma unroll
  for (int k = 0; k < 64; ++k) s = fmaf(hv[k], W[k * 256 + j], s);
#pragma unroll
  for (int k = 0; k < 64; ++k) s = fmaf(hu[k], W[(64 + k) * 256 + j], s);
  const float* efr = ef + (size_t)e * 16;
#pragma unroll
  for (int k = 0; k < 16; ++k) s = fmaf(efr[k], W[(efRow + k) * 256 + j], s);
  if (doRelu) s = s > 0.f ? s : 0.f;
  out256[t] = s;
}

// ---------------------------------------------------------------------------
// MLP1: pred_no per edge; writes out[e] = pred_no*Wl[0] + bl (base term)
// ---------------------------------------------------------------------------
__global__ __launch_bounds__(kWPB * 32) void mlp1_kernel(
    const float* __restrict__ a11,
    const __bf16* __restrict__ wfL2, const float* __restrict__ b2,
    const __bf16* __restrict__ wfL3, const float* __restrict__ b3,
    const float* __restrict__ W4,   const float* __restrict__ b4,
    const float* __restrict__ Wl,   const float* __restrict__ bl,
    float* __restrict__ out)
{
  __shared__ __align__(32) __bf16 sh_aL3[kWPB][4 * 32 * 16];
  __shared__ float                sh_a3[kWPB][16 * 64];
  const int w    = threadIdx.x >> 5;
  const int lane = threadIdx.x & 31;
  const int tile = blockIdx.x * kWPB + w;

  const int e = tile * 16 + (lane & 15);
  const float* ar  = a11 + (size_t)e * 256;
  const int    kkb = (lane >> 4) << 3;

  v16bf afrag[8];
#pragma unroll
  for (int c = 0; c < 8; ++c) {
#pragma unroll
    for (int idx = 0; idx < 16; ++idx) {
      const int kk = (idx & 7) + ((idx >> 3) << 4) + kkb;
      afrag[c][idx] = (__bf16)ar[c * 32 + kk];
    }
  }

  const float p = mlp_tail_256_128_64(afrag, wfL2, b2, wfL3, b3, W4, b4,
                                      sh_aL3[w], sh_a3[w], lane);
  if ((lane & 1) == 0) {
    const int e4 = tile * 16 + (lane >> 1);
    out[e4] = p * Wl[0] + bl[0];
  }
}

// ---------------------------------------------------------------------------
// MLP2: 622592 rows; a1 = relu(pc[e] + tz[z]); accumulate p*Wl[1+zi] into out
// ---------------------------------------------------------------------------
__global__ __launch_bounds__(kWPB * 32) void mlp2_kernel(
    const float* __restrict__ pc, const float* __restrict__ tz,
    const int* __restrict__ ei,
    const __bf16* __restrict__ wfL2, const float* __restrict__ b2,
    const __bf16* __restrict__ wfL3, const float* __restrict__ b3,
    const float* __restrict__ W4,   const float* __restrict__ b4,
    const float* __restrict__ Wl,   float* __restrict__ out)
{
  __shared__ __align__(32) __bf16 sh_aL3[kWPB][4 * 32 * 16];
  __shared__ float                sh_a3[kWPB][16 * 64];
  const int w    = threadIdx.x >> 5;
  const int lane = threadIdx.x & 31;
  const int tile = blockIdx.x * kWPB + w;

  int e, zi, zg;
  row_decode(tile * 16 + (lane & 15), ei, e, zi, zg);
  const float* pcr = pc + (size_t)e  * 256;
  const float* tzr = tz + (size_t)zg * 256;
  const int    kkb = (lane >> 4) << 3;

  v16bf afrag[8];
#pragma unroll
  for (int c = 0; c < 8; ++c) {
#pragma unroll
    for (int idx = 0; idx < 16; ++idx) {
      const int kk = (idx & 7) + ((idx >> 3) << 4) + kkb;
      const int k  = c * 32 + kk;
      float s = pcr[k] + tzr[k];
      s = s > 0.f ? s : 0.f;
      afrag[c][idx] = (__bf16)s;
    }
  }

  const float p = mlp_tail_256_128_64(afrag, wfL2, b2, wfL3, b3, W4, b4,
                                      sh_aL3[w], sh_a3[w], lane);
  if ((lane & 1) == 0) {
    int e4, zi4, zg4;
    row_decode(tile * 16 + (lane >> 1), ei, e4, zi4, zg4);
    atomicAdd(&out[e4], p * Wl[1 + zi4]);
  }
}

// ---------------------------------------------------------------------------
extern "C" void kernel_launch(void* const* d_in, const int* in_sizes, int n_in,
                              void* d_out, int out_size, void* d_ws, size_t ws_size,
                              hipStream_t stream) {
  (void)in_sizes; (void)n_in; (void)out_size; (void)ws_size;

  const float* x    = (const float*)d_in[0];
  const int*   ei   = (const int*)  d_in[1];   // [2,E]: v row then u row
  const float* ef   = (const float*)d_in[2];
  const float* Wm   = (const float*)d_in[5];
  const float* bm   = (const float*)d_in[6];
  const float* Wr   = (const float*)d_in[7];
  const float* br   = (const float*)d_in[8];
  const float* W1_1 = (const float*)d_in[9];   // [144,256]
  const float* b1_1 = (const float*)d_in[10];
  const float* W1_2 = (const float*)d_in[11];  // [256,128]
  const float* b1_2 = (const float*)d_in[12];
  const float* W1_3 = (const float*)d_in[13];  // [128,64]
  const float* b1_3 = (const float*)d_in[14];
  const float* W1_4 = (const float*)d_in[15];  // [64,1]
  const float* b1_4 = (const float*)d_in[16];
  const float* W2_1 = (const float*)d_in[17];  // [208,256]
  const float* b2_1 = (const float*)d_in[18];
  const float* W2_2 = (const float*)d_in[19];
  const float* b2_2 = (const float*)d_in[20];
  const float* W2_3 = (const float*)d_in[21];
  const float* b2_3 = (const float*)d_in[22];
  const float* W2_4 = (const float*)d_in[23];
  const float* b2_4 = (const float*)d_in[24];
  const float* Wl   = (const float*)d_in[25];
  const float* bl   = (const float*)d_in[26];
  float* out = (float*)d_out;

  // --- workspace carve (256B aligned) ---
  char* ws = (char*)d_ws;
  auto carve = [&](size_t bytes) {
    void* p = ws;
    ws += (bytes + 255) & ~(size_t)255;
    return p;
  };
  float*  agg    = (float*) carve((size_t)kNB * 64 * 4);
  float*  cnt    = (float*) carve((size_t)kNB * 4);
  float*  h      = (float*) carve((size_t)kNB * 64 * 4);
  float*  tz     = (float*) carve((size_t)kNB * 256 * 4);
  float*  pc     = (float*) carve((size_t)kE * 256 * 4);
  float*  a11    = (float*) carve((size_t)kE * 256 * 4);
  __bf16* wf1_l2 = (__bf16*)carve((size_t)256 * 128 * 2);
  __bf16* wf1_l3 = (__bf16*)carve((size_t)128 * 64 * 2);
  __bf16* wf2_l2 = (__bf16*)carve((size_t)256 * 128 * 2);
  __bf16* wf2_l3 = (__bf16*)carve((size_t)128 * 64 * 2);

  hipMemsetAsync(agg, 0, (size_t)kNB * 64 * 4, stream);
  hipMemsetAsync(cnt, 0, (size_t)kNB * 4, stream);

  wfrag_kernel<<<(256 * 128 + 255) / 256, 256, 0, stream>>>(W1_2, wf1_l2, 256, 128);
  wfrag_kernel<<<(128 * 64 + 255) / 256, 256, 0, stream>>>(W1_3, wf1_l3, 128, 64);
  wfrag_kernel<<<(256 * 128 + 255) / 256, 256, 0, stream>>>(W2_2, wf2_l2, 256, 128);
  wfrag_kernel<<<(128 * 64 + 255) / 256, 256, 0, stream>>>(W2_3, wf2_l3, 128, 64);

  conv_msg_kernel<<<(kE * 64) / 256, 256, 0, stream>>>(x, ei, ef, Wm, bm, agg, cnt);
  node_h_kernel<<<(kNB * 64) / 256, 256, 0, stream>>>(x, Wr, br, agg, cnt, h);
  tz_kernel<<<(kNB * 256) / 256, 256, 0, stream>>>(h, W2_1, tz);
  edge_affine_kernel<<<(kE * 256) / 256, 256, 0, stream>>>(h, ei, ef, W2_1, b2_1, 192, 0, pc);
  edge_affine_kernel<<<(kE * 256) / 256, 256, 0, stream>>>(h, ei, ef, W1_1, b1_1, 128, 1, a11);

  // MLP1: E/16 = 1024 tiles, 4 waves/block
  mlp1_kernel<<<(kE / 16) / kWPB, kWPB * 32, 0, stream>>>(
      a11, wf1_l2, b1_2, wf1_l3, b1_3, W1_4, b1_4, Wl, bl, out);

  // MLP2: E*Z/16 = 38912 tiles, 4 waves/block
  mlp2_kernel<<<((kE * kZ) / 16) / kWPB, kWPB * 32, 0, stream>>>(
      pc, tz, ei, wf2_l2, b2_2, wf2_l3, b2_3, W2_4, b2_4, Wl, out);
}
